// SparseMultiHeadAttention_7258494730563
// MI455X (gfx1250) — compile-verified, matrix operand order checked by reading, not run on hardware
//
#include <hip/hip_runtime.h>
#include <hip/hip_bf16.h>

#define NTOK 65536
#define CCH  512
#define NH   8
#define HD   64
#define WIN  256
#define NWIN 256
#define C3   1536

typedef __attribute__((ext_vector_type(16))) _Float16 v16h;
typedef __attribute__((ext_vector_type(8)))  _Float16 v8h;
typedef __attribute__((ext_vector_type(4)))  _Float16 v4h;
typedef __attribute__((ext_vector_type(8)))  float    v8f;

__device__ __forceinline__ v8f vzero8() {
    v8f z;
#pragma unroll
    for (int i = 0; i < 8; ++i) z[i] = 0.0f;
    return z;
}

// Load one 16x32 f16 A/B fragment from (padded) row-major LDS tile.
// ISA layout: lanes 0-15 row m=lane, halves K0-7 (v[0..7]) and K16-23 (v[8..15]);
// lanes 16-31 row m=lane-16, K8-15 and K24-31.
__device__ __forceinline__ v16h load_frag(const _Float16* p, int stride, int lane) {
    const int hh = (lane >> 4) & 1;
    const int m  = lane & 15;
    const _Float16* r = p + m * stride + hh * 8;
    v8h lo = *(const v8h*)(r);
    v8h hi = *(const v8h*)(r + 16);
    v16h out;
#pragma unroll
    for (int i = 0; i < 8; ++i) { out[i] = lo[i]; out[i + 8] = hi[i]; }
    return out;
}

__device__ __forceinline__ v8f wmma_f16(v16h a, v16h b, v8f c) {
    return __builtin_amdgcn_wmma_f32_16x16x32_f16(false, a, false, b, (short)0, c, false, false);
}

// ---------------------------------------------------------------------------
// Kernel A: qkv = x @ qkv_w^T + qkv_b, scattered to [win][head][row][64] f16
// grid (512, 12), block 256 (8 waves), tile 128x128, K-step 64
// ---------------------------------------------------------------------------
__global__ __launch_bounds__(256) void qkv_gemm_kernel(
    const float* __restrict__ x, const float* __restrict__ w,
    const float* __restrict__ b,
    _Float16* __restrict__ qo, _Float16* __restrict__ ko, _Float16* __restrict__ vo)
{
    __shared__ _Float16 As[128 * 72];
    __shared__ _Float16 Bs[128 * 72];
    const int tid  = threadIdx.x;
    const int lane = tid & 31, wv = tid >> 5;
    const int wr = wv >> 2, wc = wv & 3;           // 2x4 wave grid: 64x32 per wave
    const int br = blockIdx.x, bc = blockIdx.y;
    const int lanen = lane & 15, laneh = lane >> 4;

    v8f acc[4][2];
#pragma unroll
    for (int jr = 0; jr < 4; ++jr)
#pragma unroll
        for (int jc = 0; jc < 2; ++jc) acc[jr][jc] = vzero8();

    for (int kk = 0; kk < CCH / 64; ++kk) {
        const float* xg = x + (size_t)(br * 128) * CCH + kk * 64;
        const float* wg = w + (size_t)(bc * 128) * CCH + kk * 64;
#pragma unroll
        for (int i = 0; i < 8; ++i) {
            int linear = tid + 256 * i;            // 128 rows * 16 float4-chunks
            int r = linear >> 4, c4 = (linear & 15) * 4;
            float4 va = *(const float4*)(xg + r * CCH + c4);
            float4 vb = *(const float4*)(wg + r * CCH + c4);
            *(v4h*)(&As[r * 72 + c4]) = (v4h){(_Float16)va.x, (_Float16)va.y,
                                              (_Float16)va.z, (_Float16)va.w};
            *(v4h*)(&Bs[r * 72 + c4]) = (v4h){(_Float16)vb.x, (_Float16)vb.y,
                                              (_Float16)vb.z, (_Float16)vb.w};
        }
        __syncthreads();
#pragma unroll
        for (int kb = 0; kb < 2; ++kb) {
            v16h af[4], bf[2];
#pragma unroll
            for (int jr = 0; jr < 4; ++jr)
                af[jr] = load_frag(As + (wr * 64 + jr * 16) * 72 + kb * 32, 72, lane);
#pragma unroll
            for (int jc = 0; jc < 2; ++jc)
                bf[jc] = load_frag(Bs + (wc * 32 + jc * 16) * 72 + kb * 32, 72, lane);
#pragma unroll
            for (int jr = 0; jr < 4; ++jr)
#pragma unroll
                for (int jc = 0; jc < 2; ++jc)
                    acc[jr][jc] = wmma_f16(af[jr], bf[jc], acc[jr][jc]);
        }
        __syncthreads();
    }

#pragma unroll
    for (int jc = 0; jc < 2; ++jc) {
        const int col0 = bc * 128 + wc * 32 + jc * 16;       // 16-aligned, one head slice
        const float bias = b[col0 + lanen];
        const int s  = col0 >> 9;
        const int wi = col0 & 511;
        const int h  = wi >> 6;
        const int d0 = wi & 63;
        _Float16* base = (s == 0) ? qo : ((s == 1) ? ko : vo);
#pragma unroll
        for (int jr = 0; jr < 4; ++jr) {
            const int row0 = br * 128 + wr * 64 + jr * 16;
#pragma unroll
            for (int v = 0; v < 8; ++v) {
                const int token = row0 + v + 8 * laneh;
                const int win = token >> 8, rr = token & 255;
                const float val = acc[jr][jc][v] + bias;
                base[(size_t)(((win * NH + h) * WIN + rr)) * HD + d0 + lanen] = (_Float16)val;
            }
        }
    }
}

// ---------------------------------------------------------------------------
// Kernel B: in-place RMS norm of q,k rows (64 elems) with gamma folded in.
// grid (2048, 2), block 256; one thread per (win,head,row)
// ---------------------------------------------------------------------------
__global__ __launch_bounds__(256) void rms_norm_kernel(
    _Float16* __restrict__ q, _Float16* __restrict__ k,
    const float* __restrict__ gq, const float* __restrict__ gk)
{
    const int i = blockIdx.x * 256 + threadIdx.x;            // row id in [0, NWIN*NH*WIN)
    _Float16* base = (blockIdx.y == 0) ? q : k;
    const float* g = (blockIdx.y == 0) ? gq : gk;
    const int h = (i >> 8) & (NH - 1);
    _Float16* row = base + (size_t)i * HD;
    v8h ch[8];
    float ss = 0.0f;
#pragma unroll
    for (int c = 0; c < 8; ++c) {
        ch[c] = *(const v8h*)(row + c * 8);
#pragma unroll
        for (int j = 0; j < 8; ++j) { float f = (float)ch[c][j]; ss += f * f; }
    }
    const float inv = 1.0f / fmaxf(sqrtf(ss), 1e-12f);
#pragma unroll
    for (int c = 0; c < 8; ++c) {
        v8h o;
#pragma unroll
        for (int j = 0; j < 8; ++j)
            o[j] = (_Float16)((float)ch[c][j] * inv * g[h * HD + c * 8 + j]);
        *(v8h*)(row + c * 8) = o;
    }
}

// ---------------------------------------------------------------------------
// Kernel C: windowed attention, one workgroup per (window, head), 16 waves.
// S = Q K^T * 8 (gamma/sqrt(D) prefolded), softmax, O = P V.
// ---------------------------------------------------------------------------
__global__ __launch_bounds__(512) void attention_kernel(
    const _Float16* __restrict__ q, const _Float16* __restrict__ k,
    const _Float16* __restrict__ v, _Float16* __restrict__ o)
{
    extern __shared__ char smem[];
    _Float16* Qs = (_Float16*)smem;               // 256 x 72
    _Float16* Ks = Qs + 256 * 72;                 // 256 x 72
    _Float16* Vt = Ks + 256 * 72;                 // 64  x 264 (transposed V)
    _Float16* Ps = Vt + 64 * 264;                 // 256 x 264

    const int win = blockIdx.x, h = blockIdx.y;
    const int tid = threadIdx.x, lane = tid & 31, wv = tid >> 5;
    const int lanen = lane & 15, laneh = lane >> 4;
    const size_t base = (size_t)(win * NH + h) * WIN * HD;

#pragma unroll
    for (int i = 0; i < 4; ++i) {                 // 2048 16B chunks / 512 threads
        int ch = tid + 512 * i;
        int r = ch >> 3, c8 = (ch & 7) * 8;
        *(v8h*)(Qs + r * 72 + c8) = *(const v8h*)(q + base + r * HD + c8);
        *(v8h*)(Ks + r * 72 + c8) = *(const v8h*)(k + base + r * HD + c8);
        v8h vv = *(const v8h*)(v + base + r * HD + c8);
#pragma unroll
        for (int j = 0; j < 8; ++j) Vt[(c8 + j) * 264 + r] = vv[j];
    }
    __syncthreads();

    const int m0 = wv * 16;                       // this wave's 16 query rows
    v8f sc[16];
    {
        v16h aq[2];
#pragma unroll
        for (int kb = 0; kb < 2; ++kb)
            aq[kb] = load_frag(Qs + m0 * 72 + kb * 32, 72, lane);
#pragma unroll
        for (int t = 0; t < 16; ++t) {
            v8f c = vzero8();
#pragma unroll
            for (int kb = 0; kb < 2; ++kb) {
                v16h bk = load_frag(Ks + (t * 16) * 72 + kb * 32, 72, lane);
                c = wmma_f16(aq[kb], bk, c);
            }
            sc[t] = c;
        }
    }

    // scale = gamma-folded sqrt(D) = 8
#pragma unroll
    for (int t = 0; t < 16; ++t)
#pragma unroll
        for (int e = 0; e < 8; ++e) sc[t][e] *= 8.0f;

    float inv[8];
#pragma unroll
    for (int vv_ = 0; vv_ < 8; ++vv_) {
        float mx = -3.0e38f;
#pragma unroll
        for (int t = 0; t < 16; ++t) mx = fmaxf(mx, sc[t][vv_]);
#pragma unroll
        for (int off = 1; off < 16; off <<= 1) mx = fmaxf(mx, __shfl_xor(mx, off, 32));
        const int row = m0 + vv_ + 8 * laneh;
        float s = 0.0f;
#pragma unroll
        for (int t = 0; t < 16; ++t) {
            float p = __expf(sc[t][vv_] - mx);
            s += p;
            Ps[row * 264 + t * 16 + lanen] = (_Float16)p;
        }
#pragma unroll
        for (int off = 1; off < 16; off <<= 1) s += __shfl_xor(s, off, 32);
        inv[vv_] = 1.0f / s;
    }
    __syncthreads();

    v8f oc[4];
#pragma unroll
    for (int j = 0; j < 4; ++j) oc[j] = vzero8();
#pragma unroll
    for (int kb = 0; kb < 8; ++kb) {
        v16h ap = load_frag(Ps + m0 * 264 + kb * 32, 264, lane);
#pragma unroll
        for (int j = 0; j < 4; ++j) {
            v16h bv = load_frag(Vt + (j * 16) * 264 + kb * 32, 264, lane);
            oc[j] = wmma_f16(ap, bv, oc[j]);
        }
    }

#pragma unroll
    for (int j = 0; j < 4; ++j)
#pragma unroll
        for (int vv_ = 0; vv_ < 8; ++vv_) {
            const int row = m0 + vv_ + 8 * laneh;
            const int token = win * WIN + row;
            o[(size_t)token * CCH + h * HD + j * 16 + lanen] =
                (_Float16)(oc[j][vv_] * inv[vv_]);
        }
}

// ---------------------------------------------------------------------------
// Kernel D: out = o @ out_w^T + out_b (f16 in, f32 out). grid (512, 4).
// ---------------------------------------------------------------------------
__global__ __launch_bounds__(256) void out_gemm_kernel(
    const _Float16* __restrict__ xo, const float* __restrict__ w,
    const float* __restrict__ b, float* __restrict__ out)
{
    __shared__ _Float16 As[128 * 72];
    __shared__ _Float16 Bs[128 * 72];
    const int tid  = threadIdx.x;
    const int lane = tid & 31, wv = tid >> 5;
    const int wr = wv >> 2, wc = wv & 3;
    const int br = blockIdx.x, bc = blockIdx.y;
    const int lanen = lane & 15, laneh = lane >> 4;

    v8f acc[4][2];
#pragma unroll
    for (int jr = 0; jr < 4; ++jr)
#pragma unroll
        for (int jc = 0; jc < 2; ++jc) acc[jr][jc] = vzero8();

    for (int kk = 0; kk < CCH / 64; ++kk) {
        const _Float16* xg = xo + (size_t)(br * 128) * CCH + kk * 64;
        const float*    wg = w  + (size_t)(bc * 128) * CCH + kk * 64;
#pragma unroll
        for (int i = 0; i < 4; ++i) {             // A: 128 rows * 8 v8h-chunks
            int linear = tid + 256 * i;
            int r = linear >> 3, c8 = (linear & 7) * 8;
            *(v8h*)(As + r * 72 + c8) = *(const v8h*)(xg + r * CCH + c8);
        }
#pragma unroll
        for (int i = 0; i < 8; ++i) {             // B: 128 rows * 16 float4-chunks
            int linear = tid + 256 * i;
            int r = linear >> 4, c4 = (linear & 15) * 4;
            float4 vb = *(const float4*)(wg + r * CCH + c4);
            *(v4h*)(&Bs[r * 72 + c4]) = (v4h){(_Float16)vb.x, (_Float16)vb.y,
                                              (_Float16)vb.z, (_Float16)vb.w};
        }
        __syncthreads();
#pragma unroll
        for (int kb = 0; kb < 2; ++kb) {
            v16h af[4], bf[2];
#pragma unroll
            for (int jr = 0; jr < 4; ++jr)
                af[jr] = load_frag(As + (wr * 64 + jr * 16) * 72 + kb * 32, 72, lane);
#pragma unroll
            for (int jc = 0; jc < 2; ++jc)
                bf[jc] = load_frag(Bs + (wc * 32 + jc * 16) * 72 + kb * 32, 72, lane);
#pragma unroll
            for (int jr = 0; jr < 4; ++jr)
#pragma unroll
                for (int jc = 0; jc < 2; ++jc)
                    acc[jr][jc] = wmma_f16(af[jr], bf[jc], acc[jr][jc]);
        }
        __syncthreads();
    }

#pragma unroll
    for (int jc = 0; jc < 2; ++jc) {
        const int col0 = bc * 128 + wc * 32 + jc * 16;
        const float bias = b[col0 + lanen];
#pragma unroll
        for (int jr = 0; jr < 4; ++jr) {
            const int row0 = br * 128 + wr * 64 + jr * 16;
#pragma unroll
            for (int v = 0; v < 8; ++v) {
                const int token = row0 + v + 8 * laneh;
                out[(size_t)token * CCH + col0 + lanen] = acc[jr][jc][v] + bias;
            }
        }
    }
}

extern "C" void kernel_launch(void* const* d_in, const int* in_sizes, int n_in,
                              void* d_out, int out_size, void* d_ws, size_t ws_size,
                              hipStream_t stream) {
    const float* x       = (const float*)d_in[0];
    const float* qkv_w   = (const float*)d_in[1];
    const float* qkv_b   = (const float*)d_in[2];
    const float* q_gamma = (const float*)d_in[3];
    const float* k_gamma = (const float*)d_in[4];
    const float* out_w   = (const float*)d_in[5];
    const float* out_b   = (const float*)d_in[6];
    float* out = (float*)d_out;

    const size_t SZ = (size_t)NTOK * CCH;         // 33.5M f16 elements per buffer
    _Float16* qh = (_Float16*)d_ws;
    _Float16* kh = qh + SZ;
    _Float16* vh = kh + SZ;
    _Float16* oh = vh + SZ;

    // 1) QKV projection + bias, scattered to attention layout (f16)
    qkv_gemm_kernel<<<dim3(NTOK / 128, C3 / 128), 256, 0, stream>>>(
        x, qkv_w, qkv_b, qh, kh, vh);

    // 2) in-place RMS norm of q,k (gamma folded, sqrt(D) folded into score scale)
    rms_norm_kernel<<<dim3((NTOK * NH) / 256, 2), 256, 0, stream>>>(
        qh, kh, q_gamma, k_gamma);

    // 3) windowed attention, one WG per (window, head)
    const size_t smem = (size_t)(2 * 256 * 72 + 64 * 264 + 256 * 264) * sizeof(_Float16);
    attention_kernel<<<dim3(NWIN, NH), 512, smem, stream>>>(qh, kh, vh, oh);

    // 4) output projection -> f32
    out_gemm_kernel<<<dim3(NTOK / 128, CCH / 128), 256, 0, stream>>>(
        oh, out_w, out_b, out);
}
